// GGNN_81157702025501
// MI455X (gfx1250) — matrix operand order repaired and checked by reading, not compiled, not run on hardware
//
#include <hip/hip_runtime.h>
#include <cmath>

#define N_NODES 100000
#define N_EDGES 1600000
#define HID 128
#define N_ETYPES 4
#define N_STEPS 4

typedef __bf16 bf16;
typedef bf16  v4bf  __attribute__((ext_vector_type(4)));
typedef bf16  v8bf  __attribute__((ext_vector_type(8)));
typedef bf16  v16bf __attribute__((ext_vector_type(16)));
typedef float v8f   __attribute__((ext_vector_type(8)));
typedef int   v4i   __attribute__((vector_size(16)));

#define AS_GLOBAL __attribute__((address_space(1)))
#define AS_SHARED __attribute__((address_space(3)))

#if __has_builtin(__builtin_amdgcn_global_load_async_to_lds_b128)
#define HAVE_ASYNC_LDS 1
#else
#define HAVE_ASYNC_LDS 0
#endif

#if HAVE_ASYNC_LDS
__device__ __forceinline__ void async_wait0() {
#if __has_builtin(__builtin_amdgcn_s_wait_asynccnt)
  __builtin_amdgcn_s_wait_asynccnt(0);
#else
  asm volatile("s_wait_asynccnt 0x0" ::: "memory");
#endif
}
#endif

// ---------------------------------------------------------------------------
// WMMA fragment helpers (bf16 16x16x32, wave32 layouts per CDNA5 ISA 7.12.2)
// ---------------------------------------------------------------------------

__device__ __forceinline__ v16bf frag_cat(v8bf lo, v8bf hi) {
  return __builtin_shufflevector(lo, hi, 0,1,2,3,4,5,6,7,8,9,10,11,12,13,14,15);
}

// A fragment (16x32, row-major bf16 source with given element stride).
// lane l: row = l&15 ; k-halves at kb and kb+16 where kb = (l<16 ? 0 : 8).
__device__ __forceinline__ v16bf load_afrag_lds(const bf16* base, int stride, int lane) {
  int row = lane & 15;
  int kb  = (lane < 16) ? 0 : 8;
  const bf16* r = base + row * stride + kb;
  v8bf lo = *(const v8bf*)(r);
  v8bf hi = *(const v8bf*)(r + 16);
  return frag_cat(lo, hi);
}

// B fragment (32x16): B[k][n] = W[n][k], W row-major bf16 with element stride.
// lane l: col = l&15 ; k-halves at kb and kb+16 where kb = (l<16 ? 0 : 8).
__device__ __forceinline__ v16bf load_bfrag(const bf16* __restrict__ wbase,
                                            int stride, int lane) {
  int col = lane & 15;
  int kb  = (lane < 16) ? 0 : 8;
  const bf16* r = wbase + col * stride + kb;
  v8bf lo = *(const v8bf*)(r);
  v8bf hi = *(const v8bf*)(r + 16);
  return frag_cat(lo, hi);
}

__device__ __forceinline__ v8f wmma_bf16(v16bf a, v16bf b, v8f c) {
  return __builtin_amdgcn_wmma_f32_16x16x32_bf16(false, a, false, b,
                                                 (short)0, c, false, false);
}

__device__ __forceinline__ float sigmoidf(float x) {
  return 1.0f / (1.0f + __expf(-x));
}

// ---------------------------------------------------------------------------
// One-time prep: edge-type argmax + weight conversion to bf16
// ---------------------------------------------------------------------------
__global__ __launch_bounds__(256) void k_prep(
    const float* __restrict__ edge_feat, int* __restrict__ etype,
    const float* __restrict__ W_edge, const float* __restrict__ W_ih,
    const float* __restrict__ W_hh,
    bf16* __restrict__ We, bf16* __restrict__ Wih, bf16* __restrict__ Whh)
{
  const int idx = blockIdx.x * 256 + threadIdx.x;
  if (idx < N_EDGES) {
    const float* f = edge_feat + (size_t)idx * N_ETYPES;
    float best = f[0]; int bi = 0;
#pragma unroll
    for (int t = 1; t < N_ETYPES; ++t) {
      float v = f[t];
      if (v > best) { best = v; bi = t; }
    }
    etype[idx] = bi;
  }
  if (idx < N_ETYPES * HID * HID) We[idx] = (bf16)W_edge[idx];
  if (idx < 3 * HID * HID) {
    Wih[idx] = (bf16)W_ih[idx];
    Whh[idx] = (bf16)W_hh[idx];
  }
}

// ---------------------------------------------------------------------------
// Zero the aggregation buffer (float4 stores)
// ---------------------------------------------------------------------------
__global__ __launch_bounds__(256) void k_zero(float4* __restrict__ a, int n4)
{
  int i = blockIdx.x * 256 + threadIdx.x;
  if (i < n4) a[i] = make_float4(0.f, 0.f, 0.f, 0.f);
}

// ---------------------------------------------------------------------------
// trans[t] = bf16( h @ W_edge[t]^T + b_edge[t] )   (bf16 WMMA, f32 accum)
// trans kept in bf16: 102MB total -> L2-resident for the random edge gather.
// block: 256 thr = 8 waves, 64 node rows x 128 cols; grid (ceil(N/64), T).
// W tile (32KB bf16) staged global->LDS via async-copy with padded rows;
// h tile converted f32->bf16 into LDS. One uniform full/partial branch.
// ---------------------------------------------------------------------------
__global__ __launch_bounds__(256) void k_gemm_edge(
    const float* __restrict__ h_in, const bf16* __restrict__ We,
    const float* __restrict__ b_edge, bf16* __restrict__ trans)
{
  __shared__ bf16 sh[64 * 136];            // h tile (padded stride)
  __shared__ bf16 sw[128 * 136];           // W_edge[t] tile (padded stride)
  const int t     = blockIdx.y;
  const int node0 = blockIdx.x * 64;
  const int tid   = threadIdx.x;
  const bool full = (node0 + 64 <= N_NODES);

  // ---- stage W_edge[t]: 128x128 bf16, 2048 chunks of 16B, 8 per thread ----
  const bf16* wg = We + (size_t)t * (HID * HID);
#if HAVE_ASYNC_LDS
#pragma unroll
  for (int i = tid; i < 2048; i += 256) {
    int r = i >> 4, c = (i & 15) * 8;
    __builtin_amdgcn_global_load_async_to_lds_b128(
        (AS_GLOBAL v4i*)(wg + r * HID + c),
        (AS_SHARED v4i*)(&sw[r * 136 + c]), 0, 0);
  }
#else
  for (int i = tid; i < 2048; i += 256) {
    int r = i >> 4, c = (i & 15) * 8;
    *(v8bf*)&sw[r * 136 + c] = *(const v8bf*)(wg + r * HID + c);
  }
#endif

  // ---- stage h tile with f32->bf16 conversion ----
  if (full) {
    for (int i = tid; i < 64 * 128; i += 256) {
      int r = i >> 7, c = i & 127;
      sh[r * 136 + c] = (bf16)h_in[(size_t)(node0 + r) * HID + c];
    }
  } else {
    for (int i = tid; i < 64 * 128; i += 256) {
      int r = i >> 7, c = i & 127;
      int n = node0 + r;
      if (n >= N_NODES) n = N_NODES - 1;
      sh[r * 136 + c] = (bf16)h_in[(size_t)n * HID + c];
    }
  }
#if HAVE_ASYNC_LDS
  async_wait0();
#endif
  __syncthreads();

  const int wid  = tid >> 5;
  const int lane = tid & 31;
  const int col  = lane & 15;
  const int mb   = wid >> 1;               // M block 0..3
  const int nb0  = (wid & 1) * 4;          // N block range start

  v16bf afr[4];
#pragma unroll
  for (int k = 0; k < 4; ++k)
    afr[k] = load_afrag_lds(sh + (mb * 16) * 136 + k * 32, 136, lane);

  for (int nbi = 0; nbi < 4; ++nbi) {
    const int nb = nb0 + nbi;
    v8f acc = {};
#pragma unroll
    for (int k = 0; k < 4; ++k) {
      v16bf bfr = load_bfrag(sw + (nb * 16) * 136 + k * 32, 136, lane);
      acc = wmma_bf16(afr[k], bfr, acc);
    }
    const int o = nb * 16 + col;
    const float bias = b_edge[t * HID + o];
    bf16* dst = trans + ((size_t)t * N_NODES + node0 + mb * 16) * HID + o;
    if (full) {
#pragma unroll
      for (int i = 0; i < 8; ++i) {
        int row = (lane < 16) ? i : i + 8;
        dst[(size_t)row * HID] = (bf16)(acc[i] + bias);
      }
    } else {
#pragma unroll
      for (int i = 0; i < 8; ++i) {
        int row = (lane < 16) ? i : i + 8;
        if (node0 + mb * 16 + row < N_NODES)
          dst[(size_t)row * HID] = (bf16)(acc[i] + bias);
      }
    }
  }
}

// ---------------------------------------------------------------------------
// per-edge gather from bf16 trans[etype,src] + f32 scatter-add to a[dst]
// one wave per edge; lane handles 4 columns (8B bf16 load + 4 f32 atomics);
// trans (102MB) and a (51MB) are both L2-resident on the 192MB L2.
// ---------------------------------------------------------------------------
__global__ __launch_bounds__(256) void k_scatter(
    const bf16* __restrict__ trans, const int* __restrict__ edge_index,
    const int* __restrict__ etype, float* __restrict__ a)
{
  const int tid  = threadIdx.x;
  const int e    = blockIdx.x * 8 + (tid >> 5);
  const int lane = tid & 31;
  if (e >= N_EDGES) return;
  const int t = etype[e];
  const int s = edge_index[e];
  const int d = edge_index[N_EDGES + e];
  const v4bf m =
      *(const v4bf*)(trans + ((size_t)t * N_NODES + s) * HID + lane * 4);
  float* ap = a + (size_t)d * HID + lane * 4;
  atomicAdd(ap + 0, (float)m[0]);
  atomicAdd(ap + 1, (float)m[1]);
  atomicAdd(ap + 2, (float)m[2]);
  atomicAdd(ap + 3, (float)m[3]);
}

// ---------------------------------------------------------------------------
// Fused GRU cell: gi = a@W_ih^T + b_ih, gh = h@W_hh^T + b_hh, gates, h'
// block: 256 thr = 8 waves, 64 node rows; wave w owns hidden col block w,
// loops 4 M blocks; 6 WMMA accumulator chains per tile (r/z/n x {i,h}).
// a/h tiles staged once into padded bf16 LDS (shared by all 8 waves).
// ---------------------------------------------------------------------------
__global__ __launch_bounds__(256) void k_gru(
    const float* __restrict__ agg, const float* __restrict__ h_in,
    const bf16* __restrict__ Wih, const bf16* __restrict__ Whh,
    const float* __restrict__ b_ih, const float* __restrict__ b_hh,
    float* __restrict__ h_out)
{
  __shared__ bf16 sa[64 * 136];
  __shared__ bf16 shh[64 * 136];
  const int node0 = blockIdx.x * 64;
  const int tid   = threadIdx.x;
  const bool full = (node0 + 64 <= N_NODES);

  if (full) {
    for (int i = tid; i < 64 * 128; i += 256) {
      int r = i >> 7, c = i & 127;
      size_t g = (size_t)(node0 + r) * HID + c;
      sa[r * 136 + c]  = (bf16)agg[g];
      shh[r * 136 + c] = (bf16)h_in[g];
    }
  } else {
    for (int i = tid; i < 64 * 128; i += 256) {
      int r = i >> 7, c = i & 127;
      int n = node0 + r;
      if (n >= N_NODES) n = N_NODES - 1;
      size_t g = (size_t)n * HID + c;
      sa[r * 136 + c]  = (bf16)agg[g];
      shh[r * 136 + c] = (bf16)h_in[g];
    }
  }
  __syncthreads();

  const int wid  = tid >> 5;
  const int lane = tid & 31;
  const int cb   = wid;                // hidden column block 0..7
  const int col  = lane & 15;
  const int oc   = cb * 16 + col;      // hidden column owned by this lane

  const float bir = b_ih[oc],           bhr = b_hh[oc];
  const float biz = b_ih[HID + oc],     bhz = b_hh[HID + oc];
  const float bin = b_ih[2 * HID + oc], bhn = b_hh[2 * HID + oc];

  for (int mb = 0; mb < 4; ++mb) {
    v16bf aA[4], aH[4];
#pragma unroll
    for (int k = 0; k < 4; ++k) {
      aA[k] = load_afrag_lds(sa  + (mb * 16) * 136 + k * 32, 136, lane);
      aH[k] = load_afrag_lds(shh + (mb * 16) * 136 + k * 32, 136, lane);
    }
    v8f cir = {}, ciz = {}, cin = {};
    v8f chr_ = {}, chz = {}, chn = {};
#pragma unroll
    for (int k = 0; k < 4; ++k) {
      v16bf b;
      b = load_bfrag(Wih + (size_t)(0 * HID + cb * 16) * HID + k * 32, HID, lane);
      cir = wmma_bf16(aA[k], b, cir);
      b = load_bfrag(Wih + (size_t)(1 * HID + cb * 16) * HID + k * 32, HID, lane);
      ciz = wmma_bf16(aA[k], b, ciz);
      b = load_bfrag(Wih + (size_t)(2 * HID + cb * 16) * HID + k * 32, HID, lane);
      cin = wmma_bf16(aA[k], b, cin);
      b = load_bfrag(Whh + (size_t)(0 * HID + cb * 16) * HID + k * 32, HID, lane);
      chr_ = wmma_bf16(aH[k], b, chr_);
      b = load_bfrag(Whh + (size_t)(1 * HID + cb * 16) * HID + k * 32, HID, lane);
      chz = wmma_bf16(aH[k], b, chz);
      b = load_bfrag(Whh + (size_t)(2 * HID + cb * 16) * HID + k * 32, HID, lane);
      chn = wmma_bf16(aH[k], b, chn);
    }
    if (full) {
#pragma unroll
      for (int i = 0; i < 8; ++i) {
        int row = mb * 16 + ((lane < 16) ? i : i + 8);
        size_t idx = (size_t)(node0 + row) * HID + oc;
        float r  = sigmoidf(cir[i] + bir + chr_[i] + bhr);
        float z  = sigmoidf(ciz[i] + biz + chz[i] + bhz);
        float nn = tanhf(cin[i] + bin + r * (chn[i] + bhn));
        float ho = h_in[idx];
        h_out[idx] = (1.0f - z) * nn + z * ho;
      }
    } else {
#pragma unroll
      for (int i = 0; i < 8; ++i) {
        int row = mb * 16 + ((lane < 16) ? i : i + 8);
        int n = node0 + row;
        if (n < N_NODES) {
          size_t idx = (size_t)n * HID + oc;
          float r  = sigmoidf(cir[i] + bir + chr_[i] + bhr);
          float z  = sigmoidf(ciz[i] + biz + chz[i] + bhz);
          float nn = tanhf(cin[i] + bin + r * (chn[i] + bhn));
          float ho = h_in[idx];
          h_out[idx] = (1.0f - z) * nn + z * ho;
        }
      }
    }
  }
}

// ---------------------------------------------------------------------------
// Host orchestration
// ---------------------------------------------------------------------------
extern "C" void kernel_launch(void* const* d_in, const int* in_sizes, int n_in,
                              void* d_out, int out_size, void* d_ws, size_t ws_size,
                              hipStream_t stream) {
  const float* node_feat  = (const float*)d_in[0];
  const int*   edge_index = (const int*)d_in[1];
  const float* edge_feat  = (const float*)d_in[2];
  const float* W_edge     = (const float*)d_in[3];
  const float* b_edge     = (const float*)d_in[4];
  const float* W_ih       = (const float*)d_in[5];
  const float* W_hh       = (const float*)d_in[6];
  const float* b_ih       = (const float*)d_in[7];
  const float* b_hh       = (const float*)d_in[8];
  float* out = (float*)d_out;

  // workspace layout (bytes)
  char* ws = (char*)d_ws;
  bf16*  trans = (bf16*) (ws);                      // 4*N*128*2   = 102,400,000
  float* agg   = (float*)(ws + 102400000ULL);       // N*128*4     =  51,200,000
  float* hA    = (float*)(ws + 153600000ULL);       //             =  51,200,000
  float* hB    = (float*)(ws + 204800000ULL);       //             =  51,200,000
  int*   et    = (int*)  (ws + 256000000ULL);       // E*4         =   6,400,000
  bf16*  We    = (bf16*) (ws + 262400000ULL);       // 4*128*128*2 =     131,072
  bf16*  Wih   = (bf16*) (ws + 262531072ULL);       // 384*128*2   =      98,304
  bf16*  Whh   = (bf16*) (ws + 262629376ULL);       // 384*128*2   =      98,304

  k_prep<<<(N_EDGES + 255) / 256, 256, 0, stream>>>(
      edge_feat, et, W_edge, W_ih, W_hh, We, Wih, Whh);

  const int n4 = N_NODES * HID / 4;
  const float* h = node_feat;
  float* outs[N_STEPS] = { hA, hB, hA, out };

  for (int s = 0; s < N_STEPS; ++s) {
    k_zero<<<(n4 + 255) / 256, 256, 0, stream>>>((float4*)agg, n4);

    dim3 g1((N_NODES + 63) / 64, N_ETYPES);
    k_gemm_edge<<<g1, 256, 0, stream>>>(h, We, b_edge, trans);

    k_scatter<<<(N_EDGES + 7) / 8, 256, 0, stream>>>(trans, edge_index, et, agg);

    k_gru<<<(N_NODES + 63) / 64, 256, 0, stream>>>(
        agg, h, Wih, Whh, b_ih, b_hh, outs[s]);

    h = outs[s];
  }
}
